// MMDLoss_49795850829937
// MI455X (gfx1250) — matrix operand order repaired and checked by reading, not compiled, not run on hardware
//
#include <hip/hip_runtime.h>
#include <math.h>

// ---------------------------------------------------------------------------
// MMD loss for MI455X (gfx1250, wave32, WMMA).
//
// loss depends only on G = N * N^T where N = l2norm(concat(src,tgt)) (8192x512).
// d2 = max(2 - 2G, 0);  sum(d2) = 2n^2 - 2*||colsum(N)||^2  (no Gram pass).
// Multi-bandwidth kernel sum: with x = exp(-d2/(16*bw0)),
//   sum_{i=0..4} exp(-d2/(bw0*2^i)) = x + x^2 + x^4 + x^8 + x^16.
//
// Main kernel: f16 WMMA (v_wmma_f32_16x16x32_f16), 128x128 block/WG,
// double-buffered LDS K=32 slices (1 barrier/iter, global fetch overlapped
// with WMMA), fused exp epilogue, deterministic reductions.
// ---------------------------------------------------------------------------

#define NS    4096
#define NT    4096
#define NTOT  8192
#define DIM   512
#define PITCH 40   // halves per LDS row (32 data + 8 pad); 80 B keeps 16B align
#define KSTEPS (DIM / 32)

typedef __attribute__((ext_vector_type(16))) _Float16 v16h;
typedef __attribute__((ext_vector_type(8)))  _Float16 v8h;
typedef __attribute__((ext_vector_type(8)))  float    v8f;

// ---- Kernel A: L2-normalize each row, store as f16 -------------------------
__global__ __launch_bounds__(128)
void normalize_kernel(const float* __restrict__ src,
                      const float* __restrict__ tgt,
                      _Float16* __restrict__ N) {
  __shared__ float red[128];
  const int row = blockIdx.x;
  const float* in = (row < NS) ? (src + (size_t)row * DIM)
                               : (tgt + (size_t)(row - NS) * DIM);
  const int t = threadIdx.x;
  float v0 = in[t], v1 = in[t + 128], v2 = in[t + 256], v3 = in[t + 384];
  red[t] = v0 * v0 + v1 * v1 + v2 * v2 + v3 * v3;
  __syncthreads();
  for (int s = 64; s > 0; s >>= 1) {
    if (t < s) red[t] += red[t + s];
    __syncthreads();
  }
  const float scale = 1.0f / fmaxf(sqrtf(red[0]), 1e-12f);
  _Float16* o = N + (size_t)row * DIM;
  o[t]       = (_Float16)(v0 * scale);
  o[t + 128] = (_Float16)(v1 * scale);
  o[t + 256] = (_Float16)(v2 * scale);
  o[t + 384] = (_Float16)(v3 * scale);
}

// ---- Kernel B: per-block column partial sums (deterministic, no atomics) ---
__global__ __launch_bounds__(256)
void colpart_kernel(const _Float16* __restrict__ N, float* __restrict__ colpart) {
  const int t  = threadIdx.x;             // 0..255
  const int r0 = blockIdx.x * 64;         // 128 blocks x 64 rows
  float a0 = 0.0f, a1 = 0.0f;
  for (int r = 0; r < 64; ++r) {
    const _Float16* p = N + (size_t)(r0 + r) * DIM;
    a0 += (float)p[t];
    a1 += (float)p[t + 256];
  }
  colpart[(size_t)blockIdx.x * DIM + t]       = a0;
  colpart[(size_t)blockIdx.x * DIM + t + 256] = a1;
}

// ---- Kernel C: finish colsum, compute bandwidth scalar ---------------------
__global__ __launch_bounds__(256)
void bandwidth_kernel(const float* __restrict__ colpart, float* __restrict__ bw) {
  __shared__ float red[256];
  const int t = threadIdx.x;
  float s0 = 0.0f, s1 = 0.0f;
  for (int b = 0; b < 128; ++b) {
    s0 += colpart[(size_t)b * DIM + t];
    s1 += colpart[(size_t)b * DIM + t + 256];
  }
  red[t] = s0 * s0 + s1 * s1;
  __syncthreads();
  for (int s = 128; s > 0; s >>= 1) {
    if (t < s) red[t] += red[t + s];
    __syncthreads();
  }
  if (t == 0) {
    double n = (double)NTOT;
    double sum_d2 = 2.0 * n * n - 2.0 * (double)red[0];
    if (sum_d2 < 0.0) sum_d2 = 0.0;
    double band = sum_d2 / (n * n - n) + 1e-8;
    band *= 0.25;   // / KERNEL_MUL^(KERNEL_NUM//2) = /4
    bw[0] = (float)band;
  }
}

// ---- Fragment loaders (ISA 7.12.2 layouts, wave32) -------------------------
// A 16x32 f16: lane<16 holds K 0..7 | 16..23 ; lane>=16 holds K 8..15 | 24..31
__device__ inline v16h load_fragA(const _Float16* rowbase, int lane) {
  const int r    = lane & 15;
  const int koff = (lane >> 4) * 8;
  const _Float16* p = rowbase + r * PITCH + koff;
  v8h lo = *(const v8h*)(p);
  v8h hi = *(const v8h*)(p + 16);
  v16h out;
#pragma unroll
  for (int i = 0; i < 8; ++i) { out[i] = lo[i]; out[i + 8] = hi[i]; }
  return out;
}
// B 32x16 f16: lane<16 holds K 0..7 | 16..23 ; lane>=16 holds K 8..15 | 24..31
__device__ inline v16h load_fragB(const _Float16* rowbase, int lane) {
  const int r    = lane & 15;
  const int koff = (lane >> 4) * 8;
  const _Float16* p = rowbase + r * PITCH + koff;
  v8h lo = *(const v8h*)(p);
  v8h hi = *(const v8h*)(p + 16);
  v16h out;
#pragma unroll
  for (int i = 0; i < 8; ++i) { out[i] = lo[i]; out[i + 8] = hi[i]; }
  return out;
}

// ---- Kernel D: tiled Gram + fused MMD epilogue -----------------------------
// Grid 64x64, 256 threads (8 waves). Block computes 128x128 of G.
// Wave (wr=wave>>1, wc=wave&1) owns 2x4 tiles of 16x16.
// Double-buffered LDS: global loads for chunk k+1 overlap WMMA on chunk k;
// exactly one workgroup barrier per K-iteration.
__global__ __launch_bounds__(256)
void gram_mmd_kernel(const _Float16* __restrict__ N,
                     const float* __restrict__ bwp,
                     float* __restrict__ lpart) {
  __shared__ _Float16 As[2][128 * PITCH];
  __shared__ _Float16 Bs[2][128 * PITCH];
  __shared__ float red[256];

  const int t    = threadIdx.x;
  const int lane = t & 31;
  const int wave = t >> 5;
  const int wr   = wave >> 1;           // 0..3
  const int wc   = wave & 1;            // 0..1
  const int rowBase = blockIdx.y * 128;
  const int colBase = blockIdx.x * 128;

  // Staging assignment (straight-line, uniform): thread t copies two 16B
  // segments per operand: (row r0, part p) and (row r0+64, part p).
  const int r0 = t >> 2;
  const int pp = t & 3;
  const _Float16* gA0 = N + (size_t)(rowBase + r0)      * DIM + pp * 8;
  const _Float16* gA1 = N + (size_t)(rowBase + r0 + 64) * DIM + pp * 8;
  const _Float16* gB0 = N + (size_t)(colBase + r0)      * DIM + pp * 8;
  const _Float16* gB1 = N + (size_t)(colBase + r0 + 64) * DIM + pp * 8;
  const int lo0 = r0 * PITCH + pp * 8;
  const int lo1 = (r0 + 64) * PITCH + pp * 8;

  v8f acc[2][4];
#pragma unroll
  for (int i = 0; i < 2; ++i)
#pragma unroll
    for (int j = 0; j < 4; ++j) acc[i][j] = (v8f)(0.0f);

  // Prologue: stage chunk 0 into buffer 0.
  *(v8h*)(&As[0][lo0]) = *(const v8h*)(gA0);
  *(v8h*)(&As[0][lo1]) = *(const v8h*)(gA1);
  *(v8h*)(&Bs[0][lo0]) = *(const v8h*)(gB0);
  *(v8h*)(&Bs[0][lo1]) = *(const v8h*)(gB1);
  __syncthreads();

  for (int k = 0; k < KSTEPS - 1; ++k) {
    const int buf = k & 1;
    const int kk  = (k + 1) * 32;

    // Fetch chunk k+1 from global (L2-resident) -- overlaps with WMMA below.
    v8h ra0 = *(const v8h*)(gA0 + kk);
    v8h ra1 = *(const v8h*)(gA1 + kk);
    v8h rb0 = *(const v8h*)(gB0 + kk);
    v8h rb1 = *(const v8h*)(gB1 + kk);

    // Compute on chunk k.
    v16h af[2], bf[4];
#pragma unroll
    for (int i = 0; i < 2; ++i)
      af[i] = load_fragA(&As[buf][(wr * 2 + i) * 16 * PITCH], lane);
#pragma unroll
    for (int j = 0; j < 4; ++j)
      bf[j] = load_fragB(&Bs[buf][(wc * 4 + j) * 16 * PITCH], lane);
#pragma unroll
    for (int i = 0; i < 2; ++i)
#pragma unroll
      for (int j = 0; j < 4; ++j)
        acc[i][j] = __builtin_amdgcn_wmma_f32_16x16x32_f16(
            false, af[i], false, bf[j], (short)0, acc[i][j], false, false);

    // Stage chunk k+1 into the other buffer; one barrier per iteration.
    *(v8h*)(&As[buf ^ 1][lo0]) = ra0;
    *(v8h*)(&As[buf ^ 1][lo1]) = ra1;
    *(v8h*)(&Bs[buf ^ 1][lo0]) = rb0;
    *(v8h*)(&Bs[buf ^ 1][lo1]) = rb1;
    __syncthreads();
  }

  // Final chunk (buffer (KSTEPS-1)&1), no staging afterwards.
  {
    const int buf = (KSTEPS - 1) & 1;
    v16h af[2], bf[4];
#pragma unroll
    for (int i = 0; i < 2; ++i)
      af[i] = load_fragA(&As[buf][(wr * 2 + i) * 16 * PITCH], lane);
#pragma unroll
    for (int j = 0; j < 4; ++j)
      bf[j] = load_fragB(&Bs[buf][(wc * 4 + j) * 16 * PITCH], lane);
#pragma unroll
    for (int i = 0; i < 2; ++i)
#pragma unroll
      for (int j = 0; j < 4; ++j)
        acc[i][j] = __builtin_amdgcn_wmma_f32_16x16x32_f16(
            false, af[i], false, bf[j], (short)0, acc[i][j], false, false);
  }

  // Fused epilogue: kernel-sum with trace removal and block weights.
  const float bw0    = bwp[0];
  const float negInv = -1.0f / (bw0 * 16.0f);
  const float wSame  = 1.0f / ((float)NS * (float)(NS - 1));
  const float wCross = -1.0f / ((float)NS * (float)NT);   // covers XY and YX
  const int m_off = (lane >> 4) * 8;
  const int ncol  = lane & 15;

  float accLoss = 0.0f;
#pragma unroll
  for (int i = 0; i < 2; ++i) {
    const int Rt = rowBase + (wr * 2 + i) * 16 + m_off;
#pragma unroll
    for (int j = 0; j < 4; ++j) {
      const int Cg = colBase + (wc * 4 + j) * 16 + ncol;
      const float w = ((Rt < NS) == (Cg < NS)) ? wSame : wCross;
#pragma unroll
      for (int v = 0; v < 8; ++v) {
        const int R = Rt + v;
        const float g  = acc[i][j][v];
        const float d2 = fmaxf(2.0f - 2.0f * g, 0.0f);
        const float x  = __expf(d2 * negInv);            // exp(-d2/(16*bw0))
        const float x2 = x * x, x4 = x2 * x2, x8 = x4 * x4, x16 = x8 * x8;
        const float kv = x + x2 + x4 + x8 + x16;
        accLoss += (R == Cg) ? 0.0f : w * kv;            // trace removal
      }
    }
  }

  red[t] = accLoss;
  __syncthreads();
  for (int s = 128; s > 0; s >>= 1) {
    if (t < s) red[t] += red[t + s];
    __syncthreads();
  }
  if (t == 0) lpart[blockIdx.y * 64 + blockIdx.x] = red[0];
}

// ---- Kernel E: deterministic final reduction of 4096 block partials --------
__global__ __launch_bounds__(256)
void final_reduce_kernel(const float* __restrict__ lpart, float* __restrict__ out) {
  __shared__ float red[256];
  const int t = threadIdx.x;
  float a = 0.0f;
  for (int i = 0; i < 16; ++i) a += lpart[t + i * 256];
  red[t] = a;
  __syncthreads();
  for (int s = 128; s > 0; s >>= 1) {
    if (t < s) red[t] += red[t + s];
    __syncthreads();
  }
  if (t == 0) out[0] = red[0];
}

// ---------------------------------------------------------------------------
extern "C" void kernel_launch(void* const* d_in, const int* in_sizes, int n_in,
                              void* d_out, int out_size, void* d_ws, size_t ws_size,
                              hipStream_t stream) {
  (void)in_sizes; (void)n_in; (void)out_size; (void)ws_size;
  const float* src = (const float*)d_in[0];
  const float* tgt = (const float*)d_in[1];
  float* out = (float*)d_out;

  char* ws = (char*)d_ws;
  _Float16* N      = (_Float16*)ws;                               // 8 MB
  float*    colprt = (float*)(ws + (size_t)NTOT * DIM * 2);       // 128*512 f32
  float*    bw     = colprt + 128 * DIM;                          // 1 f32
  float*    lpart  = bw + 16;                                     // 4096 f32

  normalize_kernel<<<NTOT, 128, 0, stream>>>(src, tgt, N);
  colpart_kernel<<<NTOT / 64, 256, 0, stream>>>(N, colprt);
  bandwidth_kernel<<<1, 256, 0, stream>>>(colprt, bw);
  dim3 grid(NTOT / 128, NTOT / 128);
  gram_mmd_kernel<<<grid, 256, 0, stream>>>(N, bw, lpart);
  final_reduce_kernel<<<1, 256, 0, stream>>>(lpart, out);
}
